// PccModule_32461362823785
// MI455X (gfx1250) — matrix operand order repaired and checked by reading, not compiled, not run on hardware
//
#include <hip/hip_runtime.h>
#include <hip/hip_bf16.h>

// Problem constants (match reference)
static constexpr int L = 4096;
static constexpr int N = 8;
static constexpr int E = 512;
static constexpr int S = L;      // y has shape (L, N, E)
static constexpr float EPS = 1e-6f;

typedef __attribute__((ext_vector_type(16))) __bf16 v16bf;
typedef __attribute__((ext_vector_type(8)))  __bf16 v8bf;
typedef __attribute__((ext_vector_type(8)))  float  v8f;

__device__ __forceinline__ unsigned short f2bf(float f) {
    unsigned int u = __float_as_uint(f);
    u += 0x7FFFu + ((u >> 16) & 1u);   // round-to-nearest-even
    return (unsigned short)(u >> 16);
}

// -------- fragment loaders (ISA VGPR layouts, zero repack VALU) --------
// A 16x32 bf16: lanes 0-15: elems 0..7 = A[m,k+0..7], 8..15 = A[m,k+16..23]
//               lanes 16-31: elems 0..7 = A[m,k+8..15], 8..15 = A[m,k+24..31]
union APack { struct { v8bf lo, hi; } p; v16bf v; };

__device__ __forceinline__ v16bf load_a_bf16(const unsigned short* __restrict__ row, int k, int lh) {
    const int o = k + (lh ? 8 : 0);
    APack r;
    r.p.lo = *(const v8bf*)(row + o);
    r.p.hi = *(const v8bf*)(row + o + 16);
    return r.v;
}

// B 32x16 bf16: lane = output column; lanes 0-15 hold Bt[f,k+0..7],Bt[f,k+16..23],
// lanes 16-31 hold Bt[f,k+8..15],Bt[f,k+24..31] (same k order as A).
__device__ __forceinline__ v16bf load_b_bf16(const unsigned short* __restrict__ btrow, int k, int lh) {
    const int o = k + (lh ? 8 : 0);
    APack r;
    r.p.lo = *(const v8bf*)(btrow + o);
    r.p.hi = *(const v8bf*)(btrow + o + 16);
    return r.v;
}

#define WMMA_BF16(A, B, C) \
    __builtin_amdgcn_wmma_f32_16x16x32_bf16(false, (A), false, (B), (short)0, (C), false, false)

// shared inner loop: 64x32 output tile per wave = 4x2 of 16x16 WMMA
__device__ __forceinline__ void wave_gemm(const unsigned short* __restrict__ a0,
                                          const unsigned short* __restrict__ a1,
                                          const unsigned short* __restrict__ a2,
                                          const unsigned short* __restrict__ a3,
                                          const unsigned short* __restrict__ b0,
                                          const unsigned short* __restrict__ b1,
                                          int K, int lh, v8f (&acc)[4][2], bool pf) {
    for (int k = 0; k < K; k += 32) {
        if (pf) {
            __builtin_prefetch(a0 + k + 512, 0, 1);
            __builtin_prefetch(b0 + k + 512, 0, 1);
        }
        v16bf fa0 = load_a_bf16(a0, k, lh);
        v16bf fa1 = load_a_bf16(a1, k, lh);
        v16bf fa2 = load_a_bf16(a2, k, lh);
        v16bf fa3 = load_a_bf16(a3, k, lh);
        v16bf fb0 = load_b_bf16(b0, k, lh);
        v16bf fb1 = load_b_bf16(b1, k, lh);
        acc[0][0] = WMMA_BF16(fa0, fb0, acc[0][0]);
        acc[0][1] = WMMA_BF16(fa0, fb1, acc[0][1]);
        acc[1][0] = WMMA_BF16(fa1, fb0, acc[1][0]);
        acc[1][1] = WMMA_BF16(fa1, fb1, acc[1][1]);
        acc[2][0] = WMMA_BF16(fa2, fb0, acc[2][0]);
        acc[2][1] = WMMA_BF16(fa2, fb1, acc[2][1]);
        acc[3][0] = WMMA_BF16(fa3, fb0, acc[3][0]);
        acc[3][1] = WMMA_BF16(fa3, fb1, acc[3][1]);
    }
}

// ---------------------------------------------------------------------------
__global__ void k_zero(float* __restrict__ stats) {
    int i = blockIdx.x * blockDim.x + threadIdx.x;
    if (i < 2 * E) stats[i] = 0.0f;
}

// per-feature sum / sumsq of relu(y) over all (L*N) rows
__global__ void k_stats(const float* __restrict__ y, float* __restrict__ stats) {
    const int rowsPerBlock = (L * N) / gridDim.x;
    const int r0 = blockIdx.x * rowsPerBlock;
    const int t = threadIdx.x;                 // features t and t+256
    float s0 = 0.f, q0 = 0.f, s1 = 0.f, q1 = 0.f;
    for (int r = r0; r < r0 + rowsPerBlock; ++r) {
        const float* row = y + (size_t)r * E;
        float v = fmaxf(row[t], 0.f);        s0 += v; q0 += v * v;
        float w = fmaxf(row[t + 256], 0.f);  s1 += w; q1 += w * w;
    }
    atomicAdd(&stats[t], s0);
    atomicAdd(&stats[t + 256], s1);
    atomicAdd(&stats[E + t], q0);
    atomicAdd(&stats[E + t + 256], q1);
}

// mean / inverse-std (unbiased variance, ddof=1)
__global__ void k_finalize(const float* __restrict__ stats, float* __restrict__ meanInv) {
    int e = blockIdx.x * blockDim.x + threadIdx.x;
    if (e >= E) return;
    const float M = (float)(L * N);
    float mean = stats[e] / M;
    float var = (stats[E + e] - M * mean * mean) / (M - 1.0f);
    meanInv[e] = mean;
    meanInv[E + e] = rsqrtf(var + EPS);
}

// Wx f32 -> bf16 (rows of Wx are exactly the Bt rows for GEMM1)
__global__ void k_wx(const float* __restrict__ Wx, unsigned short* __restrict__ Wxb) {
    int i = blockIdx.x * blockDim.x + threadIdx.x;
    Wxb[i] = f2bf(Wx[i]);
}

// x (L,N,E) f32 -> xb (N,L,E) bf16 (both sides coalesced in e)
__global__ void k_xb(const float* __restrict__ x, unsigned short* __restrict__ xb) {
    size_t i = (size_t)blockIdx.x * blockDim.x + threadIdx.x;   // over L*N*E
    int e = (int)(i & (E - 1));
    size_t ln = i >> 9;                  // E = 512 = 2^9
    int n = (int)(ln & (N - 1));         // N = 8
    size_t l = ln >> 3;
    xb[((size_t)n * L + l) * E + e] = f2bf(x[i]);
}

// normalize relu(y), store TRANSPOSED bf16: y1t[n][e][s]
__global__ void k_normalize(const float* __restrict__ y, const float* __restrict__ meanInv,
                            unsigned short* __restrict__ y1t) {
    __shared__ float tile[64][65];
    const int n = blockIdx.z;
    const int s0 = blockIdx.x * 64;
    const int e0 = blockIdx.y * 64;
    const int tx = threadIdx.x & 63;
    const int ty = threadIdx.x >> 6;     // 4 rows per pass
    const float mean = meanInv[e0 + tx];
    const float inv  = meanInv[E + e0 + tx];
    #pragma unroll
    for (int i = 0; i < 64; i += 4) {
        int s = s0 + ty + i;
        float v = y[((size_t)s * N + n) * E + e0 + tx];   // coalesced in e
        v = fmaxf(v, 0.f);
        tile[ty + i][tx] = (v - mean) * inv;
    }
    __syncthreads();
    #pragma unroll
    for (int i = 0; i < 64; i += 4) {
        int e = e0 + ty + i;
        y1t[((size_t)n * E + e) * S + s0 + tx] = f2bf(tile[tx][ty + i]);  // coalesced in s
    }
}

// ---------------------------------------------------------------------------
// wave/tile decomposition shared by all GEMMs:
// block = 256 thr = 8 waves arranged 2(M) x 4(F); wave tile 64x32; block tile 128x128
struct TileIdx { int mBase, fBase, lr, lh; };
__device__ __forceinline__ TileIdx tile_idx(int m0, int f0) {
    const int wave = threadIdx.x >> 5, lane = threadIdx.x & 31;
    TileIdx t;
    t.mBase = m0 + (wave & 1) * 64;
    t.fBase = f0 + (wave >> 1) * 32;
    t.lr = lane & 15;
    t.lh = lane >> 4;
    return t;
}

// xp[n,l,f] = relu( xb[n,l,:] . Wx[f,:] + bx[f] )  -> bf16 (N,L,E)
__global__ void __launch_bounds__(256, 1)
k_gemm_xp(const unsigned short* __restrict__ xb, const unsigned short* __restrict__ Wxb,
          const float* __restrict__ bx, unsigned short* __restrict__ xp) {
    const int n = blockIdx.z;
    TileIdx ti = tile_idx(blockIdx.y * 128, blockIdx.x * 128);
    v8f acc[4][2] = {};
    const unsigned short* aBase = xb + (size_t)n * L * E;
    wave_gemm(aBase + (size_t)(ti.mBase + ti.lr) * E,
              aBase + (size_t)(ti.mBase + 16 + ti.lr) * E,
              aBase + (size_t)(ti.mBase + 32 + ti.lr) * E,
              aBase + (size_t)(ti.mBase + 48 + ti.lr) * E,
              Wxb + (size_t)(ti.fBase + ti.lr) * E,
              Wxb + (size_t)(ti.fBase + 16 + ti.lr) * E,
              E, ti.lh, acc, false);
    #pragma unroll
    for (int t = 0; t < 4; ++t)
      #pragma unroll
      for (int u = 0; u < 2; ++u) {
        int col = ti.fBase + u * 16 + ti.lr;
        float bias = bx[col];
        int rowTop = ti.mBase + t * 16 + ti.lh * 8;
        #pragma unroll
        for (int r = 0; r < 8; ++r) {
            float v = fmaxf(acc[t][u][r] + bias, 0.f);
            xp[((size_t)n * L + rowTop + r) * E + col] = f2bf(v);
        }
      }
}

// yy[n,e,f] = sum_s y1t[n,e,s]*y1t[n,f,s] / (S*E)  -> bf16 (N,E,E), symmetric
__global__ void __launch_bounds__(256, 1)
k_gemm_cov(const unsigned short* __restrict__ y1t, unsigned short* __restrict__ yy) {
    const int n = blockIdx.z;
    TileIdx ti = tile_idx(blockIdx.y * 128, blockIdx.x * 128);
    v8f acc[4][2] = {};
    const unsigned short* base = y1t + (size_t)n * E * S;
    wave_gemm(base + (size_t)(ti.mBase + ti.lr) * S,
              base + (size_t)(ti.mBase + 16 + ti.lr) * S,
              base + (size_t)(ti.mBase + 32 + ti.lr) * S,
              base + (size_t)(ti.mBase + 48 + ti.lr) * S,
              base + (size_t)(ti.fBase + ti.lr) * S,
              base + (size_t)(ti.fBase + 16 + ti.lr) * S,
              S, ti.lh, acc, true);
    const float scale = 1.0f / ((float)S * (float)E);
    #pragma unroll
    for (int t = 0; t < 4; ++t)
      #pragma unroll
      for (int u = 0; u < 2; ++u) {
        int col = ti.fBase + u * 16 + ti.lr;
        int rowTop = ti.mBase + t * 16 + ti.lh * 8;
        #pragma unroll
        for (int r = 0; r < 8; ++r)
            yy[((size_t)n * E + rowTop + r) * E + col] = f2bf(acc[t][u][r] * scale);
      }
}

// out[l,n,f] = sum_e xp[n,l,e] * yy[n,e,f]   (yy symmetric -> Bt rows = yy rows)
__global__ void __launch_bounds__(256, 1)
k_gemm_out(const unsigned short* __restrict__ xp, const unsigned short* __restrict__ yy,
           float* __restrict__ out) {
    const int n = blockIdx.z;
    TileIdx ti = tile_idx(blockIdx.y * 128, blockIdx.x * 128);
    v8f acc[4][2] = {};
    const unsigned short* aBase = xp + (size_t)n * L * E;
    const unsigned short* bBase = yy + (size_t)n * E * E;
    wave_gemm(aBase + (size_t)(ti.mBase + ti.lr) * E,
              aBase + (size_t)(ti.mBase + 16 + ti.lr) * E,
              aBase + (size_t)(ti.mBase + 32 + ti.lr) * E,
              aBase + (size_t)(ti.mBase + 48 + ti.lr) * E,
              bBase + (size_t)(ti.fBase + ti.lr) * E,
              bBase + (size_t)(ti.fBase + 16 + ti.lr) * E,
              E, ti.lh, acc, false);
    #pragma unroll
    for (int t = 0; t < 4; ++t)
      #pragma unroll
      for (int u = 0; u < 2; ++u) {
        int col = ti.fBase + u * 16 + ti.lr;
        int rowTop = ti.mBase + t * 16 + ti.lh * 8;
        #pragma unroll
        for (int r = 0; r < 8; ++r)
            out[((size_t)(rowTop + r) * N + n) * E + col] = acc[t][u][r];
      }
}

// ---------------------------------------------------------------------------
extern "C" void kernel_launch(void* const* d_in, const int* in_sizes, int n_in,
                              void* d_out, int out_size, void* d_ws, size_t ws_size,
                              hipStream_t stream) {
    const float* x  = (const float*)d_in[0];   // (L,N,E)
    const float* y  = (const float*)d_in[1];   // (L,N,E)
    const float* Wx = (const float*)d_in[2];   // (E,E)
    const float* bx = (const float*)d_in[3];   // (E)
    float* out = (float*)d_out;                // (L,N,E)
    (void)in_sizes; (void)n_in; (void)out_size; (void)ws_size;

    // workspace layout (byte offsets, all 32B aligned)
    char* ws = (char*)d_ws;
    size_t off = 0;
    float* stats        = (float*)(ws + off);  off += 4096;                    // 2*E f32
    float* meanInv      = (float*)(ws + off);  off += 4096;                    // 2*E f32
    unsigned short* Wxb = (unsigned short*)(ws + off);  off += (size_t)E * E * 2;      // 512 KB
    unsigned short* xb  = (unsigned short*)(ws + off);  off += (size_t)N * L * E * 2;  // 32 MB
    unsigned short* y1t = (unsigned short*)(ws + off);  off += (size_t)N * E * S * 2;  // 32 MB
    unsigned short* xp  = (unsigned short*)(ws + off);  off += (size_t)N * L * E * 2;  // 32 MB
    unsigned short* yyb = (unsigned short*)(ws + off);  off += (size_t)N * E * E * 2;  // 4 MB

    k_zero<<<dim3(4), dim3(256), 0, stream>>>(stats);
    k_stats<<<dim3(256), dim3(256), 0, stream>>>(y, stats);
    k_finalize<<<dim3(2), dim3(256), 0, stream>>>(stats, meanInv);
    k_wx<<<dim3((E * E) / 256), dim3(256), 0, stream>>>(Wx, Wxb);
    k_xb<<<dim3((L * N * E) / 256), dim3(256), 0, stream>>>(x, xb);
    k_normalize<<<dim3(S / 64, E / 64, N), dim3(256), 0, stream>>>(y, meanInv, y1t);

    k_gemm_xp <<<dim3(E / 128, L / 128, N), dim3(256), 0, stream>>>(xb, Wxb, bx, xp);
    k_gemm_cov<<<dim3(E / 128, E / 128, N), dim3(256), 0, stream>>>(y1t, yyb);
    k_gemm_out<<<dim3(E / 128, L / 128, N), dim3(256), 0, stream>>>(xp, yyb, out);
}
